// Head_11948599017759
// MI455X (gfx1250) — matrix operand order repaired and checked by reading, not compiled, not run on hardware
//
#include <hip/hip_runtime.h>

// Problem dims (compile-time constants from the reference)
#define BATCH 4
#define SEQ   4096
#define CDIM  1024
#define HDIM  64

typedef __attribute__((ext_vector_type(16))) __bf16 v16bf;
typedef __attribute__((ext_vector_type(8)))  __bf16 v8bf;
typedef __attribute__((ext_vector_type(8)))  float  v8f;
typedef __attribute__((ext_vector_type(4)))  float  v4f;

static __device__ __forceinline__ v8f wmma_bf16(v16bf a, v16bf b, v8f c) {
  // 8 args: (neg_a, A, neg_b, B, c_mod, C, reuse_a, reuse_b)
  return __builtin_amdgcn_wmma_f32_16x16x32_bf16(false, a, false, b, (short)0, c,
                                                 false, false);
}

// CDNA5 async global->LDS copy (16B per lane), tracked by ASYNCcnt.
static __device__ __forceinline__ void async_ld16(unsigned lds_off,
                                                  const void* gaddr) {
  asm volatile("global_load_async_to_lds_b128 %0, %1, off"
               :
               : "v"(lds_off), "v"(gaddr)
               : "memory");
}
// Wait until at most N async ops remain outstanding (in-order completion).
static __device__ __forceinline__ void wait_async0() {
  asm volatile("s_wait_asynccnt 0x0" ::: "memory");
}
static __device__ __forceinline__ void wait_async1() {
  asm volatile("s_wait_asynccnt 0x1" ::: "memory");
}
static __device__ __forceinline__ void wait_async4() {
  asm volatile("s_wait_asynccnt 0x4" ::: "memory");
}
static __device__ __forceinline__ unsigned lds_off_of(const void* p) {
  // LDS aperture keeps the byte offset in addr[31:0] (ISA 10.2 aperture map).
  return (unsigned)(unsigned long long)p;
}

// ---------------------------------------------------------------------------
// Kernel 0: transpose+convert weights: Wt[w][c][k] = (bf16)W_w[k][c],
// w in {Q,K,V}, c in [0,64), k in [0,1024).  Makes every projection
// B-fragment a single contiguous 32B load.
// ---------------------------------------------------------------------------
__global__ __launch_bounds__(256)
void wt_prep_kernel(const float* __restrict__ Wq,
                    const float* __restrict__ Wk,
                    const float* __restrict__ Wv,
                    __bf16* __restrict__ Wt) {
  int idx = blockIdx.x * 256 + threadIdx.x;   // 0 .. 3*64*1024-1
  int w   = idx >> 16;
  int rem = idx & 65535;
  int c   = rem >> 10;
  int k   = rem & 1023;
  const float* W = (w == 0) ? Wq : (w == 1) ? Wk : Wv;
  Wt[idx] = (__bf16)W[k * HDIM + c];
}

// ---------------------------------------------------------------------------
// Kernel 1: fused QKV projection.  One block (4 waves) per 16-row M-tile;
// wave = 16-col N-tile.  The 16x32 fp32 x-tile (2KB) per K-step is
// double-buffered through LDS with async copies overlapping the 3 WMMAs
// (Q, K, V) against contiguous Wt B-fragments.
// Q,K stored row-major bf16 [B*T,64]; V stored transposed bf16 [B][64][T].
// ---------------------------------------------------------------------------
__global__ __launch_bounds__(128)
void qkv_proj_kernel(const float* __restrict__ x,
                     const __bf16* __restrict__ Wt,
                     __bf16* __restrict__ Qo,
                     __bf16* __restrict__ Ko,
                     __bf16* __restrict__ Vt) {
  __shared__ __align__(64) float xbuf[2][16 * 32];   // 2 x 2KB staging tiles

  const int tid   = threadIdx.x;
  const int lane  = tid & 31;
  const int wave  = tid >> 5;           // N-tile 0..3
  const int nl    = lane & 15;
  const int half  = lane >> 4;
  const int mrow0 = blockIdx.x * 16;
  const int ncol0 = wave * 16;

  const __bf16* wtq = Wt + 0 * HDIM * CDIM + (size_t)(ncol0 + nl) * CDIM;
  const __bf16* wtk = Wt + 1 * HDIM * CDIM + (size_t)(ncol0 + nl) * CDIM;
  const __bf16* wtv = Wt + 2 * HDIM * CDIM + (size_t)(ncol0 + nl) * CDIM;

  const unsigned xb_off = lds_off_of(&xbuf[0][0]);
  // Per-thread staging chunk: row = tid>>3, 16B sub-chunk = tid&7.
  const char* xsrc_base =
      (const char*)(x + (size_t)(mrow0 + (tid >> 3)) * CDIM) + (tid & 7) * 16;

  // Prologue: stage first x tile into buffer 0.
  async_ld16(xb_off + tid * 16, xsrc_base);

  v8f accQ = {}, accK = {}, accV = {};
  int buf = 0;
  for (int kk = 0; kk < CDIM; kk += 32) {
    // Issue next tile into the alternate buffer (free since its consumers
    // passed the end-of-iteration barrier of the previous step).
    const bool has_next = (kk + 32) < CDIM;
    if (has_next)
      async_ld16(xb_off + (buf ^ 1) * 2048 + tid * 16,
                 xsrc_base + (size_t)(kk + 32) * 4);
    // Wait for the current tile only (async loads retire in order).
    if (has_next) wait_async1(); else wait_async0();
    __syncthreads();

    // A fragment (16x32 bf16) from LDS: element e -> K=half*16+e.
    const float* xr = xbuf[buf] + nl * 32;
    v4f a0 = *(const v4f*)(xr + half * 16);
    v4f a1 = *(const v4f*)(xr + half * 16 + 4);
    v4f a2 = *(const v4f*)(xr + half * 16 + 8);
    v4f a3 = *(const v4f*)(xr + half * 16 + 12);
    v16bf a;
#pragma unroll
    for (int i = 0; i < 4; ++i) {
      a[i]      = (__bf16)a0[i];
      a[4 + i]  = (__bf16)a1[i];
      a[8 + i]  = (__bf16)a2[i];
      a[12 + i] = (__bf16)a3[i];
    }
    // B fragments: contiguous 32B loads from transposed weights.
    v16bf bq = *(const v16bf*)(wtq + kk + half * 16);
    v16bf bk = *(const v16bf*)(wtk + kk + half * 16);
    v16bf bv = *(const v16bf*)(wtv + kk + half * 16);
    accQ = wmma_bf16(a, bq, accQ);
    accK = wmma_bf16(a, bk, accK);
    accV = wmma_bf16(a, bv, accV);
    __syncthreads();   // all waves done with xbuf[buf]
    buf ^= 1;
  }

  // C layout: VGPR r -> row mrow0 + r + half*8, col = ncol0 + nl.
#pragma unroll
  for (int r = 0; r < 8; ++r) {
    const int row = mrow0 + r + half * 8;
    Qo[(size_t)row * HDIM + ncol0 + nl] = (__bf16)accQ[r];
    Ko[(size_t)row * HDIM + ncol0 + nl] = (__bf16)accK[r];
    const int b = row >> 12;             // / SEQ
    const int t = row & (SEQ - 1);
    Vt[(size_t)b * HDIM * SEQ + (size_t)(ncol0 + nl) * SEQ + t] =
        (__bf16)accV[r];
  }
}

// ---------------------------------------------------------------------------
// Kernel 2: flash attention, causal + padding mask.  Block = 4 waves = 64
// consecutive query rows of one batch.  Shared 32-key K/V tiles are
// double-buffered in LDS via async copies (DMA overlaps WMMA); each wave
// runs online softmax on its own 16-row tile (S and O via WMMA, P
// re-striped through a per-wave LDS bounce).
// ---------------------------------------------------------------------------
__global__ __launch_bounds__(128)
void flash_attn_kernel(const __bf16* __restrict__ Q,
                       const __bf16* __restrict__ K,
                       const __bf16* __restrict__ Vt,
                       const int* __restrict__ pmask,
                       float* __restrict__ out) {
  __shared__ __align__(64) __bf16 kbuf[2][32 * 64];    // [key][head]  2x4KB
  __shared__ __align__(64) __bf16 vbuf[2][64 * 32];    // [col][key]   2x4KB
  __shared__ __align__(64) __bf16 pbuf[4][16 * 32];    // per-wave P bounce

  const int tid   = threadIdx.x;
  const int lane  = tid & 31;
  const int wv    = tid >> 5;
  const int nl    = lane & 15;
  const int half  = lane >> 4;
  const int qt    = blockIdx.x * 4 + wv;        // query tile (same batch/block)
  const int b     = qt >> 8;                    // / (SEQ/16)
  const int qrow0 = (qt & 255) * 16;
  const size_t qbase = ((size_t)b * SEQ + qrow0) * HDIM;

  // Q A-fragments for head-dim halves kb = 0 and 32 (loaded once).
  v16bf qf[2];
#pragma unroll
  for (int h = 0; h < 2; ++h) {
    const __bf16* qp = Q + qbase + (size_t)nl * HDIM + h * 32 + half * 16;
    v8bf lo = *(const v8bf*)(qp);
    v8bf hi = *(const v8bf*)(qp + 8);
#pragma unroll
    for (int i = 0; i < 8; ++i) { qf[h][i] = lo[i]; qf[h][8 + i] = hi[i]; }
  }

  v8f o0 = {}, o1 = {}, o2 = {}, o3 = {};
  float m[8], l[8];
#pragma unroll
  for (int r = 0; r < 8; ++r) { m[r] = -1e30f; l[r] = 0.0f; }

  // Block-uniform key loop bound: last query row covered by this block.
  const int lastq_blk = (((blockIdx.x * 4 + 3) & 255) * 16) + 15;
  __bf16* pl = pbuf[wv];
  const __bf16* Kb  = K + (size_t)b * SEQ * HDIM;
  const __bf16* Vtb = Vt + (size_t)b * HDIM * SEQ;
  const int* pmb = pmask + b * SEQ;
  const unsigned kb_off = lds_off_of(&kbuf[0][0]);
  const unsigned vb_off = lds_off_of(&vbuf[0][0]);

  // Issue the 4 async ops (K: 2 contiguous 2KB halves; V: 64B per col)
  // for key tile kt0 into LDS buffer `bsel`.
  auto issue_tile = [&](int kt0, int bsel) {
    const char* ksrc = (const char*)(Kb + (size_t)kt0 * HDIM);
    const unsigned kdst = kb_off + bsel * 4096;
    async_ld16(kdst + tid * 16,        ksrc + tid * 16);
    async_ld16(kdst + 2048 + tid * 16, ksrc + 2048 + tid * 16);
    const unsigned vdst = vb_off + bsel * 4096;
#pragma unroll
    for (int i = 0; i < 2; ++i) {
      const int ch  = tid + i * 128;          // 256 chunks of 16B
      const int col = ch >> 2, sub = ch & 3;
      async_ld16(vdst + ch * 16,
                 (const char*)(Vtb + (size_t)col * SEQ + kt0) + sub * 16);
    }
  };

  issue_tile(0, 0);   // prologue
  int buf = 0;
  for (int kt0 = 0; kt0 <= lastq_blk; kt0 += 32) {
    // Pipeline: issue next tile, then wait only for this tile's 4 ops.
    const bool has_next = (kt0 + 32) <= lastq_blk;
    if (has_next) issue_tile(kt0 + 32, buf ^ 1);
    if (has_next) wait_async4(); else wait_async0();
    __syncthreads();

    const __bf16* kb = kbuf[buf];
    const __bf16* vb = vbuf[buf];

    // ---- S = Q @ K^T from LDS (two 16-key tiles, head-dim halves) ----
    v8f s0 = {}, s1 = {};
#pragma unroll
    for (int h = 0; h < 2; ++h) {
      v16bf bf0 = *(const v16bf*)(kb + (size_t)nl * HDIM + h * 32 + half * 16);
      v16bf bf1 =
          *(const v16bf*)(kb + (size_t)(nl + 16) * HDIM + h * 32 + half * 16);
      s0 = wmma_bf16(qf[h], bf0, s0);
      s1 = wmma_bf16(qf[h], bf1, s1);
    }

    // ---- mask + scale + online softmax (fp32 VALU) ----
    const int key0 = kt0 + nl, key1 = kt0 + 16 + nl;
    const int pm0 = pmb[key0];
    const int pm1 = pmb[key1];
    float p0[8], p1[8], alpha[8];
#pragma unroll
    for (int r = 0; r < 8; ++r) {
      const int row = qrow0 + r + half * 8;
      float v0 = (key0 <= row && pm0 != 0) ? s0[r] * 0.125f : -1e30f;
      float v1 = (key1 <= row && pm1 != 0) ? s1[r] * 0.125f : -1e30f;
      float rv = fmaxf(v0, v1);
#pragma unroll
      for (int msk = 1; msk < 16; msk <<= 1)
        rv = fmaxf(rv, __shfl_xor(rv, msk, 32));   // stays within 16-lane half
      const float mn = fmaxf(m[r], rv);
      alpha[r] = __expf(m[r] - mn);
      p0[r] = __expf(v0 - mn);
      p1[r] = __expf(v1 - mn);
      float rs = p0[r] + p1[r];
#pragma unroll
      for (int msk = 1; msk < 16; msk <<= 1)
        rs += __shfl_xor(rs, msk, 32);
      l[r] = l[r] * alpha[r] + rs;
      m[r] = mn;
    }
#pragma unroll
    for (int r = 0; r < 8; ++r) {
      o0[r] *= alpha[r]; o1[r] *= alpha[r];
      o2[r] *= alpha[r]; o3[r] *= alpha[r];
    }

    // ---- P: C-layout -> A-layout via per-wave LDS bounce ----
#pragma unroll
    for (int r = 0; r < 8; ++r) {
      pl[(r + half * 8) * 32 + nl]      = (__bf16)p0[r];
      pl[(r + half * 8) * 32 + nl + 16] = (__bf16)p1[r];
    }
    v8bf plo = *(const v8bf*)(pl + nl * 32 + half * 16);
    v8bf phi = *(const v8bf*)(pl + nl * 32 + half * 16 + 8);
    v16bf pf;
#pragma unroll
    for (int i = 0; i < 8; ++i) { pf[i] = plo[i]; pf[8 + i] = phi[i]; }

    // ---- O += P @ V from LDS ----
    o0 = wmma_bf16(pf, *(const v16bf*)(vb + (0 * 16 + nl) * 32 + half * 16), o0);
    o1 = wmma_bf16(pf, *(const v16bf*)(vb + (1 * 16 + nl) * 32 + half * 16), o1);
    o2 = wmma_bf16(pf, *(const v16bf*)(vb + (2 * 16 + nl) * 32 + half * 16), o2);
    o3 = wmma_bf16(pf, *(const v16bf*)(vb + (3 * 16 + nl) * 32 + half * 16), o3);

    __syncthreads();   // tile consumed by all waves; buffer reusable
    buf ^= 1;
  }

  // ---- normalize and store fp32 output ----
#pragma unroll
  for (int r = 0; r < 8; ++r) {
    const float inv = (l[r] > 0.0f) ? 1.0f / l[r] : 0.0f;
    const size_t row = (size_t)b * SEQ + qrow0 + r + half * 8;
    out[row * HDIM +  0 + nl] = o0[r] * inv;
    out[row * HDIM + 16 + nl] = o1[r] * inv;
    out[row * HDIM + 32 + nl] = o2[r] * inv;
    out[row * HDIM + 48 + nl] = o3[r] * inv;
  }
}

// ---------------------------------------------------------------------------
// Launch: inputs are {x, Wq, Wk, Wv, padding_mask}; output [B,T,64] fp32.
// Workspace: Q (2MB) | K (2MB) | Vt (2MB) | Wt (384KB), all bf16.
// ---------------------------------------------------------------------------
extern "C" void kernel_launch(void* const* d_in, const int* in_sizes, int n_in,
                              void* d_out, int out_size, void* d_ws,
                              size_t ws_size, hipStream_t stream) {
  const float* x  = (const float*)d_in[0];
  const float* Wq = (const float*)d_in[1];
  const float* Wk = (const float*)d_in[2];
  const float* Wv = (const float*)d_in[3];
  const int*   pm = (const int*)d_in[4];
  float* out = (float*)d_out;

  const size_t n = (size_t)BATCH * SEQ * HDIM;
  __bf16* Qo = (__bf16*)d_ws;
  __bf16* Ko = Qo + n;
  __bf16* Vt = Ko + n;
  __bf16* Wt = Vt + n;   // 3*64*1024 bf16

  wt_prep_kernel<<<(3 * HDIM * CDIM) / 256, 256, 0, stream>>>(Wq, Wk, Wv, Wt);

  qkv_proj_kernel<<<(BATCH * SEQ) / 16, 128, 0, stream>>>(x, Wt, Qo, Ko, Vt);

  flash_attn_kernel<<<((BATCH * SEQ) / 16) / 4, 128, 0, stream>>>(Qo, Ko, Vt,
                                                                  pm, out);
}
